// GNN_Dense_83047487635730
// MI455X (gfx1250) — compile-verified, matrix operand order checked by reading, not run on hardware
//
#include <hip/hip_runtime.h>
#include <hip/hip_bf16.h>
#include <math.h>

typedef __attribute__((ext_vector_type(16))) _Float16 v16h;
typedef __attribute__((ext_vector_type(8)))  _Float16 v8h;
typedef __attribute__((ext_vector_type(8)))  float    v8f;

#define N_NODES  50000
#define N_EDGES  1600000
#define N_GRAPHS 512
#define IN_DIM   128
#define HID      128
#define HALF     64
#define NEG_SLOPE 0.2f
#define E_TOT (N_EDGES + N_NODES)

// ---------------- utility ----------------
__global__ void fill_f32(float* __restrict__ p, float v, int n) {
  int i = blockIdx.x * blockDim.x + threadIdx.x;
  if (i < n) p[i] = v;
}

__device__ __forceinline__ void atomicMaxF(float* addr, float val) {
  // sign-split trick: int ordering matches float for >=0, reversed-uint for <0
  if (val >= 0.f) atomicMax((int*)addr, __float_as_int(val));
  else            atomicMin((unsigned int*)addr, __float_as_uint(val));
}

// ---------------- h[N,64] = x[N,128] @ W[128,64] via v_wmma_f32_16x16x32_f16 ----
// W is staged in LDS pre-permuted into WMMA B-fragment layout:
//   frag(c,t): 32 lanes x 16 contiguous halves   (c = k-chunk, t = n-tile)
//   lane (r,hi): halves[0..7]  = W[c*32 + hi*8 + i      ][t*16 + r]
//                halves[8..15] = W[c*32 + 16 + hi*8 + i ][t*16 + r]
__global__ __launch_bounds__(128)
void gemm_xw(const float* __restrict__ x, const float* __restrict__ W,
             float* __restrict__ h, int nrows) {
  __shared__ __align__(32) _Float16 Wlds[4 * 4 * 32 * 16];   // 16 KB
  const int tid = threadIdx.x;

  // fill: 1024 groups of 8 consecutive halves; each group is one 16B LDS store
  for (int g = tid; g < 1024; g += 128) {
    const int f     = g << 3;            // base half index
    const int ihalf = (f >> 3) & 1;      // 0 -> halves 0..7, 1 -> halves 8..15
    const int ln    = (f >> 4) & 31;
    const int t     = (f >> 9) & 3;
    const int c     =  f >> 11;
    const int r  = ln & 15;
    const int hi = ln >> 4;
    const int kbase = c * 32 + hi * 8 + (ihalf ? 16 : 0);
    const int col   = t * 16 + r;
    v8h tmp;
    #pragma unroll
    for (int i = 0; i < 8; ++i) tmp[i] = (_Float16)W[(kbase + i) * HALF + col];
    *(v8h*)(&Wlds[f]) = tmp;
  }
  __syncthreads();

  const int wave = tid >> 5;
  const int lane = tid & 31;
  const int r    = lane & 15;
  const int hi   = lane >> 4;                      // half-wave id (0/1)
  const int m0   = (blockIdx.x * 4 + wave) * 16;   // 16-row strip per wave
  if (m0 >= nrows) return;                         // uniform per-wave exit

  const v16h* __restrict__ Bfrag = (const v16h*)Wlds;  // [(c*4+t)*32 + lane]
  v8f acc[4] = {};                                 // 4 x (16x16) n-tiles = 16x64

  #pragma unroll
  for (int c = 0; c < 4; ++c) {
    const int k0 = c * 32;
    // A fragment: 16-bit A 16x32 layout (lane<16: K 0..7 & 16..23; lane>=16: +8)
    v16h a;
    const float* xr = x + (size_t)(m0 + r) * IN_DIM + k0;
    #pragma unroll
    for (int i = 0; i < 8; ++i) a[i]     = (_Float16)xr[hi * 8 + i];
    #pragma unroll
    for (int i = 0; i < 8; ++i) a[8 + i] = (_Float16)xr[16 + hi * 8 + i];

    #pragma unroll
    for (int t = 0; t < 4; ++t) {
      v16h b = Bfrag[(c * 4 + t) * 32 + lane];     // 2 x ds_load_b128
      acc[t] = __builtin_amdgcn_wmma_f32_16x16x32_f16(
          false, a, false, b, (short)0, acc[t], false, false);
    }
  }
  // C/D layout: VGPR j -> row m0 + hi*8 + j, col n0 + r
  #pragma unroll
  for (int t = 0; t < 4; ++t)
    #pragma unroll
    for (int j = 0; j < 8; ++j)
      h[(size_t)(m0 + hi * 8 + j) * HALF + 16 * t + r] = acc[t][j];
}

// ---------------- per-node attention scores es = h.a_src, ed = h.a_dst ---------
__global__ void node_scores(const float* __restrict__ h,
                            const float* __restrict__ a_src,
                            const float* __restrict__ a_dst,
                            float* __restrict__ es, float* __restrict__ ed) {
  int i = blockIdx.x * blockDim.x + threadIdx.x;
  if (i >= N_NODES) return;
  const float4* hr  = (const float4*)(h + (size_t)i * HALF);
  const float4* as4 = (const float4*)a_src;
  const float4* ad4 = (const float4*)a_dst;
  float s = 0.f, d = 0.f;
  #pragma unroll
  for (int j = 0; j < 16; ++j) {
    float4 v = hr[j], a = as4[j], b = ad4[j];
    s += v.x * a.x + v.y * a.y + v.z * a.z + v.w * a.w;
    d += v.x * b.x + v.y * b.y + v.z * b.z + v.w * b.w;
  }
  es[i] = s; ed[i] = d;
}

// ---------------- pass 1: raw score + LeakyReLU + segment max ------------------
__global__ void edge_pass1(const int* __restrict__ src, const int* __restrict__ dst,
                           const float* __restrict__ es, const float* __restrict__ ed,
                           float* __restrict__ score, float* __restrict__ m) {
  int e = blockIdx.x * blockDim.x + threadIdx.x;
  if (e >= E_TOT) return;
  int s, d;
  if (e < N_EDGES) { s = src[e]; d = dst[e]; } else { s = d = e - N_EDGES; }
  float v = es[s] + ed[d];
  v = (v >= 0.f) ? v : NEG_SLOPE * v;
  score[e] = v;
  atomicMaxF(&m[d], v);
}

// ---------------- pass 2: exp(score - m[d]) + segment sum ----------------------
__global__ void edge_pass2(const int* __restrict__ dst,
                           float* __restrict__ score,
                           const float* __restrict__ m, float* __restrict__ z) {
  int e = blockIdx.x * blockDim.x + threadIdx.x;
  if (e >= E_TOT) return;
  int d = (e < N_EDGES) ? dst[e] : (e - N_EDGES);
  float w = __expf(score[e] - m[d]);
  score[e] = w;                       // stash the exp weight
  atomicAdd(&z[d], w);
}

// ---------------- pass 3: out[d] += alpha * h[s]  (one wave per edge) ----------
__global__ __launch_bounds__(256)
void edge_pass3(const int* __restrict__ src, const int* __restrict__ dst,
                const float* __restrict__ score, const float* __restrict__ z,
                const float* __restrict__ h, float* __restrict__ out) {
  const int lane = threadIdx.x & 31;
  const int e = blockIdx.x * 8 + (threadIdx.x >> 5);
  if (e >= E_TOT) return;
  int s, d;
  if (e < N_EDGES) { s = src[e]; d = dst[e]; } else { s = d = e - N_EDGES; }
  float alpha = score[e] / (z[d] + 1e-16f);
  const float* hs = h + (size_t)s * HALF;
  float* od = out + (size_t)d * HALF;
  atomicAdd(&od[lane],      alpha * hs[lane]);
  atomicAdd(&od[32 + lane], alpha * hs[32 + lane]);
}

// ---------------- concat + bias + relu -> next layer input ---------------------
__global__ void combine_relu(const float* __restrict__ outA, const float* __restrict__ outB,
                             const float* __restrict__ bias, float* __restrict__ xn) {
  int idx = blockIdx.x * blockDim.x + threadIdx.x;
  if (idx >= N_NODES * HALF) return;
  int n = idx >> 6, j = idx & 63;
  float bj = bias[j];
  float a = outA[idx] + bj;
  float c = outB[idx] + bj;
  xn[(size_t)n * HID + j]        = a > 0.f ? a : 0.f;
  xn[(size_t)n * HID + HALF + j] = c > 0.f ? c : 0.f;
}

// ---------------- graph pooling ------------------------------------------------
__global__ void pool_kernel(const float* __restrict__ xc, const int* __restrict__ batch,
                            float* __restrict__ pooled) {
  int idx = blockIdx.x * blockDim.x + threadIdx.x;
  if (idx >= N_NODES * HID) return;
  int n = idx >> 7, j = idx & 127;
  atomicAdd(&pooled[(size_t)batch[n] * HID + j], xc[idx]);
}

// ---------------- final projection: one wave32 per graph -----------------------
__global__ void final_kernel(const float* __restrict__ pooled, const float* __restrict__ Wf,
                             const float* __restrict__ bf, float* __restrict__ y) {
  int g = blockIdx.x;
  int lane = threadIdx.x;
  float p = 0.f;
  for (int k = lane; k < HID; k += 32) p += pooled[(size_t)g * HID + k] * Wf[k];
  #pragma unroll
  for (int off = 16; off > 0; off >>= 1) p += __shfl_down(p, off, 32);
  if (lane == 0) y[g] = p + bf[0];
}

// ==============================================================================
extern "C" void kernel_launch(void* const* d_in, const int* in_sizes, int n_in,
                              void* d_out, int out_size, void* d_ws, size_t ws_size,
                              hipStream_t stream) {
  (void)in_sizes; (void)n_in; (void)out_size; (void)ws_size;
  const float* x0    = (const float*)d_in[0];
  const int*   eiA   = (const int*)d_in[1];      // [2, E] flat: src then dst
  const int*   eiB   = (const int*)d_in[2];
  const int*   batch = (const int*)d_in[3];
  const float* W[3]  = {(const float*)d_in[4],  (const float*)d_in[8],  (const float*)d_in[12]};
  const float* aS[3] = {(const float*)d_in[5],  (const float*)d_in[9],  (const float*)d_in[13]};
  const float* aD[3] = {(const float*)d_in[6],  (const float*)d_in[10], (const float*)d_in[14]};
  const float* bb[3] = {(const float*)d_in[7],  (const float*)d_in[11], (const float*)d_in[15]};
  const float* Wf    = (const float*)d_in[16];
  const float* bf    = (const float*)d_in[17];
  float* y = (float*)d_out;

  // workspace carve-up (256B aligned)
  char* wsp = (char*)d_ws;
  auto alloc = [&](size_t bytes) -> float* {
    float* p = (float*)wsp;
    wsp += (bytes + 255) & ~(size_t)255;
    return p;
  };
  float* xA     = alloc((size_t)N_NODES * HID * 4);
  float* xB     = alloc((size_t)N_NODES * HID * 4);
  float* hbuf   = alloc((size_t)N_NODES * HALF * 4);
  float* es     = alloc((size_t)N_NODES * 4);
  float* ed     = alloc((size_t)N_NODES * 4);
  float* mbuf   = alloc((size_t)N_NODES * 4);
  float* zbuf   = alloc((size_t)N_NODES * 4);
  float* score  = alloc((size_t)E_TOT * 4);
  float* outA   = alloc((size_t)N_NODES * HALF * 4);
  float* outB   = alloc((size_t)N_NODES * HALF * 4);
  float* pooled = alloc((size_t)N_GRAPHS * HID * 4);

  const int TB = 256;
  const int gemmBlocks = (N_NODES + 63) / 64;           // 4 waves x 16 rows / block
  const int nodeBlocks = (N_NODES + TB - 1) / TB;
  const int edgeBlocks = (E_TOT + TB - 1) / TB;
  const int aggBlocks  = (E_TOT + 7) / 8;               // 8 edges (waves) / block

  const float* xin = x0;
  float* xout = xA;
  for (int l = 0; l < 3; ++l) {
    gemm_xw<<<gemmBlocks, 128, 0, stream>>>(xin, W[l], hbuf, N_NODES);
    node_scores<<<nodeBlocks, TB, 0, stream>>>(hbuf, aS[l], aD[l], es, ed);

    for (int br = 0; br < 2; ++br) {
      const int* src = (br == 0) ? eiA : eiB;
      const int* dst = src + N_EDGES;
      float* out = (br == 0) ? outA : outB;
      fill_f32<<<nodeBlocks, TB, 0, stream>>>(mbuf, -INFINITY, N_NODES);
      fill_f32<<<nodeBlocks, TB, 0, stream>>>(zbuf, 0.f, N_NODES);
      fill_f32<<<(N_NODES * HALF + TB - 1) / TB, TB, 0, stream>>>(out, 0.f, N_NODES * HALF);
      edge_pass1<<<edgeBlocks, TB, 0, stream>>>(src, dst, es, ed, score, mbuf);
      edge_pass2<<<edgeBlocks, TB, 0, stream>>>(dst, score, mbuf, zbuf);
      edge_pass3<<<aggBlocks, TB, 0, stream>>>(src, dst, score, zbuf, hbuf, out);
    }

    combine_relu<<<(N_NODES * HALF + TB - 1) / TB, TB, 0, stream>>>(outA, outB, bb[l], xout);
    xin = xout;
    xout = (xout == xA) ? xB : xA;
  }

  fill_f32<<<(N_GRAPHS * HID + TB - 1) / TB, TB, 0, stream>>>(pooled, 0.f, N_GRAPHS * HID);
  pool_kernel<<<(N_NODES * HID + TB - 1) / TB, TB, 0, stream>>>(xin, batch, pooled);
  final_kernel<<<N_GRAPHS, 32, 0, stream>>>(pooled, Wf, bf, y);
}
